// MultiheadAttention_32100585570456
// MI455X (gfx1250) — matrix operand order repaired and checked by reading, not compiled, not run on hardware
//
#include <hip/hip_runtime.h>
#include <hip/hip_bf16.h>

#define DIMV   1024
#define SEQV   2048
#define NHEADS 16
#define HDIM   64
#define BATCHV 2
#define MROWS  (BATCHV * SEQV)   // 4096

typedef __attribute__((ext_vector_type(16))) _Float16 v16h;
typedef __attribute__((ext_vector_type(8)))  _Float16 v8h;
typedef __attribute__((ext_vector_type(8)))  float    v8f;

union HFrag { v16h v; v8h h[2]; };

// A-matrix fragment (16x32 f16, M x K): lane L -> row L%16,
// chunks at K = 8*(L/16) and 16+8*(L/16) (ISA 7.12.2 16-bit A layout).
__device__ __forceinline__ v16h load_frag_a(const _Float16* base, int ld, int lane) {
  const int m = lane & 15, half = lane >> 4;
  const _Float16* p = base + (size_t)m * ld + 8 * half;
  HFrag f;
  f.h[0] = *(const v8h*)(p);
  f.h[1] = *(const v8h*)(p + 16);
  return f.v;
}

// B-matrix fragment (32x16 f16, K x N) read from a [N, K] row-major source
// (i.e. B = srcT): lane L -> column L%16, chunks at K = 8*(L/16) and 16+8*(L/16).
__device__ __forceinline__ v16h load_frag_b(const _Float16* base, int ld, int lane) {
  const int n = lane & 15, half = lane >> 4;
  const _Float16* p = base + (size_t)n * ld + 8 * half;
  HFrag f;
  f.h[0] = *(const v8h*)(p);
  f.h[1] = *(const v8h*)(p + 16);
  return f.v;
}

__device__ __forceinline__ v8f wmma_f16(v16h a, v16h b, v8f c) {
  return __builtin_amdgcn_wmma_f32_16x16x32_f16(
      /*neg_a=*/false, a, /*neg_b=*/false, b,
      /*c_mod=*/(short)0, c, /*reuse_a=*/false, /*reuse_b=*/false);
}

// CDNA5 async tile copy: global -> LDS, 16 bytes per lane, tracked by ASYNCcnt.
// LDS byte address lives in addr[31:0] of the generic shared pointer.
__device__ __forceinline__ void async_copy_b128(const void* lds_ptr, const void* gptr) {
  const unsigned lds_addr = (unsigned)(uintptr_t)lds_ptr;
  const unsigned long long ga = (unsigned long long)(uintptr_t)gptr;
  asm volatile("global_load_async_to_lds_b128 %0, %1, off"
               :: "v"(lds_addr), "v"(ga)
               : "memory");
}
// Async loads complete in order: waiting "<= 2" retires everything except the
// 2 loads this wave just issued for the next tile (true double buffering).
__device__ __forceinline__ void wait_async_le2() {
  asm volatile("s_wait_asynccnt 2" ::: "memory");
}
__device__ __forceinline__ void wait_async0() {
  asm volatile("s_wait_asynccnt 0" ::: "memory");
}

// ---------------------------------------------------------------- convert
__global__ __launch_bounds__(256) void cvt_f32_f16(const float* __restrict__ in,
                                                   _Float16* __restrict__ out, int n) {
  const int i = blockIdx.x * blockDim.x + threadIdx.x;
  if (i < n) out[i] = (_Float16)in[i];
}

// ---------------------------------------------------------------- GEMM
// C[M,N] = A[M,K] * W[N,K]^T   (torch Linear convention), M=4096, N=K=1024.
// 256 threads = 8 waves; block tile 64x64; A/B K-step tiles (64x32 f16, 4 KB
// each) staged into LDS with double-buffered GLOBAL_LOAD_ASYNC_TO_LDS_B128.
// Pipeline per 32-K step: issue next tile async, s_wait_asynccnt 2 (current
// tile resident, next still in flight), barrier, WMMA from LDS, barrier.
// mode 0: f16 out, [B,H,S,HD]     (Q, K)
// mode 1: f16 out, [B,H,HD,S]     (V transposed)
// mode 2: f32 out, [M,N] + bias   (output projection)
__global__ __launch_bounds__(256) void gemm_wmma(
    const _Float16* __restrict__ A,
    const _Float16* __restrict__ W,
    _Float16* __restrict__ OutH,
    float* __restrict__ OutF,
    const float* __restrict__ bias,
    int mode) {
  const int K = DIMV, N = DIMV;
  __shared__ _Float16 sA[2][64 * 32];
  __shared__ _Float16 sB[2][64 * 32];

  const int tid = threadIdx.x, lane = tid & 31, wave = tid >> 5;
  const int rg = wave & 3;        // row group  (16 rows of the 64-row tile)
  const int cg = wave >> 2;       // col group  (32 cols of the 64-col tile)
  const int mB = blockIdx.x * 64;
  const int nB = blockIdx.y * 64;

  // async-copy mapping: thread -> (row 0..63, 16B chunk 0..3) of a 64x32 tile
  const int crow = tid >> 2;
  const int ccol = (tid & 3) * 8;
  const _Float16* gA = A + (size_t)(mB + crow) * K + ccol;
  const _Float16* gB = W + (size_t)(nB + crow) * K + ccol;
  _Float16* lA0 = &sA[0][crow * 32 + ccol];
  _Float16* lB0 = &sB[0][crow * 32 + ccol];
  _Float16* lA1 = &sA[1][crow * 32 + ccol];
  _Float16* lB1 = &sB[1][crow * 32 + ccol];

  // fragment bases
  const _Float16* fA0 = &sA[0][16 * rg * 32];
  const _Float16* fA1 = &sA[1][16 * rg * 32];
  const _Float16* fB0a = &sB[0][(cg * 32) * 32];
  const _Float16* fB0b = &sB[0][(cg * 32 + 16) * 32];
  const _Float16* fB1a = &sB[1][(cg * 32) * 32];
  const _Float16* fB1b = &sB[1][(cg * 32 + 16) * 32];

  // prologue: stage tile 0 into buffer 0
  async_copy_b128(lA0, gA);
  async_copy_b128(lB0, gB);

  v8f acc0 = {}, acc1 = {};
  for (int k = 0; k < K; k += 64) {
    // -------- phase 0: compute tile k (buf 0), prefetch tile k+32 (buf 1)
    async_copy_b128(lA1, gA + k + 32);
    async_copy_b128(lB1, gB + k + 32);
    wait_async_le2();          // my buf0 chunks retired (in-order completion)
    __syncthreads();           // everyone's buf0 resident
    {
      const v16h a  = load_frag_a(fA0, 32, lane);
      const v16h b0 = load_frag_b(fB0a, 32, lane);
      const v16h b1 = load_frag_b(fB0b, 32, lane);
      acc0 = wmma_f16(a, b0, acc0);
      acc1 = wmma_f16(a, b1, acc1);
    }
    __syncthreads();           // everyone done reading buf0

    // -------- phase 1: compute tile k+32 (buf 1), prefetch tile k+64 (buf 0)
    if (k + 64 < K) {
      async_copy_b128(lA0, gA + k + 64);
      async_copy_b128(lB0, gB + k + 64);
      wait_async_le2();
    } else {
      wait_async0();           // nothing new issued: must drain fully
    }
    __syncthreads();           // everyone's buf1 resident
    {
      const v16h a  = load_frag_a(fA1, 32, lane);
      const v16h b0 = load_frag_b(fB1a, 32, lane);
      const v16h b1 = load_frag_b(fB1b, 32, lane);
      acc0 = wmma_f16(a, b0, acc0);
      acc1 = wmma_f16(a, b1, acc1);
    }
    __syncthreads();           // everyone done reading buf1
  }

  // C/D layout: VGPR r -> M = r + 8*(lane/16), N = lane%16.
  const int m0 = mB + rg * 16;
  const int n0 = nB + cg * 32;
  const int half8 = (lane >> 4) << 3;
  const int nc = lane & 15;
#pragma unroll
  for (int r = 0; r < 8; ++r) {
    const int m  = m0 + r + half8;
    const int e0 = n0 + nc, e1 = n0 + 16 + nc;
    const float v0 = acc0[r], v1 = acc1[r];
    if (mode == 2) {
      OutF[(size_t)m * N + e0] = v0 + bias[e0];
      OutF[(size_t)m * N + e1] = v1 + bias[e1];
    } else {
      const int bb = m >> 11, s = m & (SEQV - 1);
      const int h0 = e0 >> 6, d0 = e0 & 63;
      const int h1 = e1 >> 6, d1 = e1 & 63;
      size_t i0, i1;
      if (mode == 0) {
        i0 = ((((size_t)bb * NHEADS + h0) * SEQV) + s) * HDIM + d0;
        i1 = ((((size_t)bb * NHEADS + h1) * SEQV) + s) * HDIM + d1;
      } else {
        i0 = (((size_t)bb * NHEADS + h0) * HDIM + d0) * SEQV + s;
        i1 = (((size_t)bb * NHEADS + h1) * HDIM + d1) * SEQV + s;
      }
      OutH[i0] = (_Float16)v0;
      OutH[i1] = (_Float16)v1;
    }
  }
}

// ---------------------------------------------------------------- attention
// One workgroup per (batch, head, 64-query block). 8 waves:
//   wave w: rg = w&3 -> 16 query rows; cg = w>>2 in {0,1}.
// Scores: wave computes S tile (rows 16*rg, kv cols 16*cg) of a 64x32 block.
// Ctx:    wave owns tiles (rg, 2*cg) and (rg, 2*cg+1) over head-dim 64.
// Online softmax staged through LDS (also performs the C/D -> A transpose).
__global__ __launch_bounds__(256) void attn_flash(
    const _Float16* __restrict__ Q,   // [B,H,S,HD]
    const _Float16* __restrict__ Kh,  // [B,H,S,HD]
    const _Float16* __restrict__ Vt,  // [B,H,HD,S]
    _Float16* __restrict__ Ctx) {     // [B*S, D]
  __shared__ float    sS[64 * 32];
  __shared__ _Float16 sP[64 * 32];
  __shared__ float    sM[64], sL[64], sA[64];

  const int tid  = threadIdx.x;
  const int lane = tid & 31;
  const int wave = tid >> 5;
  const int rg = wave & 3;
  const int cg = wave >> 2;
  const int q0 = blockIdx.x * 64;
  const int h  = blockIdx.y;
  const int b  = blockIdx.z;
  const size_t bh = (size_t)b * NHEADS + h;

  if (tid < 64) { sM[tid] = -__builtin_inff(); sL[tid] = 0.0f; }
  __syncthreads();

  // Q fragments for this wave's row group stay in registers (K = 0..31, 32..63).
  const _Float16* qbase = Q + ((bh * SEQV) + q0 + 16 * rg) * HDIM;
  const v16h aq0 = load_frag_a(qbase,      HDIM, lane);
  const v16h aq1 = load_frag_a(qbase + 32, HDIM, lane);

  v8f c0 = {}, c1 = {};
  const int dg0 = cg * 2, dg1 = cg * 2 + 1;
  const int kv_end = q0 + 64;            // causal: tokens <= last query row

  for (int j0 = 0; j0 < kv_end; j0 += 32) {
    // ---- scores: S = Q * K^T (head-dim 64 = 2 WMMAs)
    const _Float16* kbase = Kh + ((bh * SEQV) + j0 + 16 * cg) * HDIM;
    const v16h bk0 = load_frag_b(kbase,      HDIM, lane);
    const v16h bk1 = load_frag_b(kbase + 32, HDIM, lane);
    v8f s = {};
    s = wmma_f16(aq0, bk0, s);
    s = wmma_f16(aq1, bk1, s);
    {
      const int col   = 16 * cg + (lane & 15);
      const int rbase = 16 * rg + ((lane >> 4) << 3);
#pragma unroll
      for (int r = 0; r < 8; ++r) sS[(rbase + r) * 32 + col] = s[r];
    }
    __syncthreads();

    // ---- online softmax, one thread per query row
    if (tid < 64) {
      const int qrow = q0 + tid;
      const float mold = sM[tid];
      float mnew = mold;
#pragma unroll
      for (int c = 0; c < 32; ++c) {
        float v = sS[tid * 32 + c] * 0.125f;          // 1/sqrt(64)
        if (j0 + c > qrow) v = -__builtin_inff();     // causal mask
        sS[tid * 32 + c] = v;
        mnew = fmaxf(mnew, v);
      }
      const float alpha = __expf(mold - mnew);
      float lsum = 0.0f;
#pragma unroll
      for (int c = 0; c < 32; ++c) {
        const float p = __expf(sS[tid * 32 + c] - mnew);
        sP[tid * 32 + c] = (_Float16)p;
        lsum += p;
      }
      sL[tid] = sL[tid] * alpha + lsum;
      sM[tid] = mnew;
      sA[tid] = alpha;
    }
    __syncthreads();

    // ---- rescale running ctx, then ctx += P * V
    {
      const int rbase = 16 * rg + ((lane >> 4) << 3);
#pragma unroll
      for (int r = 0; r < 8; ++r) {
        const float al = sA[rbase + r];
        c0[r] *= al;
        c1[r] *= al;
      }
    }
    const v16h pa = load_frag_a(&sP[16 * rg * 32], 32, lane);   // 16x32 f16 from LDS
    const _Float16* vbase = Vt + (bh * HDIM) * SEQV + j0;
    const v16h vb0 = load_frag_b(vbase + (size_t)(16 * dg0) * SEQV, SEQV, lane);
    const v16h vb1 = load_frag_b(vbase + (size_t)(16 * dg1) * SEQV, SEQV, lane);
    c0 = wmma_f16(pa, vb0, c0);
    c1 = wmma_f16(pa, vb1, c1);
    __syncthreads();
  }

  // ---- finalize: divide by l, write ctx as [B*S, D] f16
  {
    const int rbase = 16 * rg + ((lane >> 4) << 3);
    const int n = lane & 15;
#pragma unroll
    for (int r = 0; r < 8; ++r) {
      const int row  = rbase + r;
      const float inv = 1.0f / sL[row];
      const size_t orow = ((size_t)b * SEQV + q0 + row) * DIMV + (size_t)h * HDIM;
      Ctx[orow + 16 * dg0 + n] = (_Float16)(c0[r] * inv);
      Ctx[orow + 16 * dg1 + n] = (_Float16)(c1[r] * inv);
    }
  }
}

// ---------------------------------------------------------------- launch
extern "C" void kernel_launch(void* const* d_in, const int* in_sizes, int n_in,
                              void* d_out, int out_size, void* d_ws, size_t ws_size,
                              hipStream_t stream) {
  const float* x  = (const float*)d_in[0];
  const float* Wq = (const float*)d_in[1];
  const float* Wk = (const float*)d_in[2];
  const float* Wv = (const float*)d_in[3];
  const float* Wo = (const float*)d_in[4];
  const float* bo = (const float*)d_in[5];

  char* ws = (char*)d_ws;
  const size_t MB = 1u << 20;
  _Float16* xh  = (_Float16*)(ws + 0 * MB);    // 8 MB  [4096,1024]
  _Float16* wqh = (_Float16*)(ws + 8 * MB);    // 2 MB
  _Float16* wkh = (_Float16*)(ws + 10 * MB);
  _Float16* wvh = (_Float16*)(ws + 12 * MB);
  _Float16* woh = (_Float16*)(ws + 14 * MB);
  _Float16* Qh  = (_Float16*)(ws + 16 * MB);   // 8 MB  [B,H,S,64]
  _Float16* Kx  = (_Float16*)(ws + 24 * MB);   // 8 MB  [B,H,S,64]
  _Float16* Vt  = (_Float16*)(ws + 32 * MB);   // 8 MB  [B,H,64,S]
  _Float16* Ctx = (_Float16*)(ws + 40 * MB);   // 8 MB  [4096,1024]

  const int nX = MROWS * DIMV, nW = DIMV * DIMV;
  cvt_f32_f16<<<(nX + 255) / 256, 256, 0, stream>>>(x, xh, nX);
  cvt_f32_f16<<<(nW + 255) / 256, 256, 0, stream>>>(Wq, wqh, nW);
  cvt_f32_f16<<<(nW + 255) / 256, 256, 0, stream>>>(Wk, wkh, nW);
  cvt_f32_f16<<<(nW + 255) / 256, 256, 0, stream>>>(Wv, wvh, nW);
  cvt_f32_f16<<<(nW + 255) / 256, 256, 0, stream>>>(Wo, woh, nW);

  dim3 ggrid(MROWS / 64, DIMV / 64);
  gemm_wmma<<<ggrid, 256, 0, stream>>>(xh, wqh, Qh, nullptr, nullptr, 0);
  gemm_wmma<<<ggrid, 256, 0, stream>>>(xh, wkh, Kx, nullptr, nullptr, 0);
  gemm_wmma<<<ggrid, 256, 0, stream>>>(xh, wvh, Vt, nullptr, nullptr, 1);

  dim3 agrid(SEQV / 64, NHEADS, BATCHV);
  attn_flash<<<agrid, 256, 0, stream>>>(Qh, Kx, Vt, Ctx);

  gemm_wmma<<<ggrid, 256, 0, stream>>>(Ctx, woh, nullptr, (float*)d_out, bo, 2);
}
